// TransformerBlock_16767552324053
// MI455X (gfx1250) — compile-verified, matrix operand order checked by reading, not run on hardware
//
#include <hip/hip_runtime.h>
#include <hip/hip_bf16.h>
#include <math.h>

// ---------------------------------------------------------------------------
// Single-token transformer block, L=65536, D=512, H=4, HD=128.
//
// Roofline: mandatory traffic = read x once = 128 MB -> ~5.5us @ 23.3 TB/s.
// Exact algebraic refactor removes the 34-GFLOP K/V GEMMs:
//   s_h[l]   = xn_l . qk_h + c_h,   qk_h = W_k,h^T q_h   (precomputed)
//   o_h      = W_v,h . t_h + b_v,h, t_h  = sum_l softmax_h[l] * xn_l
// Main pass (k_main): per 64-row tile -> LayerNorm -> f16 in LDS ->
// WMMA scores (transposed: S^T = QK * XN^T, both operands are contiguous
// LDS rows -> ds_load_b128 fragment loads) -> local softmax -> WMMA
// weighted-xn accumulation -> per-block flash-softmax partials.
// ---------------------------------------------------------------------------

typedef _Float16 v16h __attribute__((ext_vector_type(16)));
typedef _Float16 v8h  __attribute__((ext_vector_type(8)));
typedef float    v8f  __attribute__((ext_vector_type(8)));

#define LSEQ 65536
#define DDIM 512
#define TILE 64
#define NBLK (LSEQ / TILE)          // 1024 tiles
#define INV_SQRT_HD 0.08838834764831845f   // 1/sqrt(128)

// workspace layout (float offsets)
#define WS_Q    0                   // 512   : q row
#define WS_QK   512                 // 2048  : qk[h][d], pre-scaled
#define WS_SB   2560                // 4     : per-head score bias (scaled)
#define WS_MG   2576                // 4     : global max per head
#define WS_LG   2580                // 4     : global denom per head
#define WS_OPRE 2592                // 2048  : normalized weighted xn, [h][d]
#define WS_MP   8192                // 4096  : per-block max   [blk][h]
#define WS_LP   12288               // 4096  : per-block denom [blk][h]
#define WS_TP   16384               // NBLK*2048 : per-block t partial [blk][h][d]
// total ~8.4 MB of f32 workspace

// dynamic LDS layout for k_main (bytes); all row strides keep 16B alignment
#define SM_XNH   0                  // _Float16 [64][520]  = 66560 B
#define SM_QKH   66560              // _Float16 [16][520]  = 16640 B (rows>=4 zero)
#define SM_PM    83200              // _Float16 [16][72]   =  2304 B
#define SM_SC    85504              // float    [64][4]    =  1024 B
#define SM_MH    86528              // float[4]
#define SM_LH    86544              // float[4]
#define SM_SB    86560              // float[4]
#define SMEM_BYTES 86576

// Fragment loader: ISA 16-bit A-matrix layout for 16x32 chunk.
// Lane (m, hi) holds k = hi*8..hi*8+7 and k = 16+hi*8..16+hi*8+7, which are
// two contiguous, 16B-aligned 8-half runs of one row -> two ds_load_b128.
// The same register image serves as the B-fragment of the row-matrix^T.
__device__ __forceinline__ v16h load_frag(const _Float16* base, int hi) {
  const v8h* p = (const v8h*)(base + hi * 8);
  v8h lo = p[0];                    // k = hi*8 .. hi*8+7
  v8h hb = p[2];                    // k = 16+hi*8 .. 16+hi*8+7
  return __builtin_shufflevector(lo, hb, 0, 1, 2, 3, 4, 5, 6, 7,
                                 8, 9, 10, 11, 12, 13, 14, 15);
}

// ---------------------------------------------------------------------------
// K0: LN(row0), q = xn0 @ Wq^T + bq, qk_h = Wk,h^T q_h (scaled), score bias.
// ---------------------------------------------------------------------------
__global__ __launch_bounds__(256) void k_prep(const float* __restrict__ x,
                                              const float* __restrict__ w_in,
                                              const float* __restrict__ b_in,
                                              const float* __restrict__ g1,
                                              const float* __restrict__ be1,
                                              float* __restrict__ ws) {
  __shared__ float xn0[512];
  __shared__ float qsh[512];
  __shared__ float redS[256], redQ[256];
  const int t = threadIdx.x;

  float a0 = x[t], a1 = x[t + 256];
  redS[t] = a0 + a1;
  redQ[t] = a0 * a0 + a1 * a1;
  __syncthreads();
  for (int s = 128; s > 0; s >>= 1) {
    if (t < s) { redS[t] += redS[t + s]; redQ[t] += redQ[t + s]; }
    __syncthreads();
  }
  const float mean = redS[0] * (1.0f / 512.0f);
  const float var  = redQ[0] * (1.0f / 512.0f) - mean * mean;
  const float rstd = rsqrtf(var + 1e-5f);
  xn0[t]       = (a0 - mean) * rstd * g1[t]       + be1[t];
  xn0[t + 256] = (a1 - mean) * rstd * g1[t + 256] + be1[t + 256];
  __syncthreads();

  // q[j] = xn0 . w_in[j,:] + b_in[j]
  for (int j = t; j < 512; j += 256) {
    const float* wr = w_in + (size_t)j * 512;
    float acc = b_in[j];
    for (int d = 0; d < 512; ++d) acc += xn0[d] * wr[d];
    qsh[j] = acc;
    ws[WS_Q + j] = acc;
  }
  __syncthreads();

  // qk[h][d] = (1/sqrt(HD)) * sum_i q[h*128+i] * Wk[h*128+i][d]
  for (int e = t; e < 2048; e += 256) {          // e = h*512 + d
    const int h = e >> 9, d = e & 511;
    const float* wr = w_in + (size_t)(512 + h * 128) * 512 + d;
    float acc = 0.0f;
    for (int i = 0; i < 128; ++i) acc += qsh[h * 128 + i] * wr[(size_t)i * 512];
    ws[WS_QK + e] = acc * INV_SQRT_HD;
  }
  if (t < 4) {
    float acc = 0.0f;
    for (int i = 0; i < 128; ++i) acc += qsh[t * 128 + i] * b_in[512 + t * 128 + i];
    ws[WS_SB + t] = acc * INV_SQRT_HD;
  }
}

// ---------------------------------------------------------------------------
// K1: main streaming pass over x. One block = 64 rows.
// ---------------------------------------------------------------------------
__global__ __launch_bounds__(256) void k_main(const float* __restrict__ x,
                                              const float* __restrict__ g1,
                                              const float* __restrict__ be1,
                                              float* __restrict__ ws) {
  extern __shared__ char smem[];
  _Float16* xnh  = (_Float16*)(smem + SM_XNH);   // [64][520] f16 (pad 8)
  _Float16* qkh  = (_Float16*)(smem + SM_QKH);   // [16][520] f16, rows>=4 zero
  _Float16* pmat = (_Float16*)(smem + SM_PM);    // [16][72]  f16, rows>=4 zero
  float*    sc   = (float*)   (smem + SM_SC);    // [64][4]
  float*    mh   = (float*)   (smem + SM_MH);
  float*    lh   = (float*)   (smem + SM_LH);
  float*    sbl  = (float*)   (smem + SM_SB);

  const int tid  = threadIdx.x;
  const int blk  = blockIdx.x;
  const int lane = tid & 31;
  const int wv   = tid >> 5;

  // stage 0: stage qk rows (f16, zero-padded to 16 heads) + score bias.
  // row index is wave-uniform per iteration (256 | 512), so the branch is
  // uniform: no per-lane EXEC masking.
  for (int e = tid; e < 16 * 512; e += 256) {
    const int row = e >> 9, col = e & 511;
    qkh[row * 520 + col] =
        (row < 4) ? (_Float16)ws[WS_QK + row * 512 + col] : (_Float16)0.0f;
  }
  if (tid < 4) sbl[tid] = ws[WS_SB + tid];

  // stage 1: LayerNorm of 64 rows -> xnh (f16). 4 threads per row.
  {
    const int row = tid >> 2;
    const int q4  = tid & 3;
    const float* xr = x + (size_t)(blk * TILE + row) * DDIM + q4 * 128;
    __builtin_prefetch(xr, 0, 0);                // global_prefetch_b8
    float4 buf[32];
    const float4* x4 = (const float4*)xr;
    float s = 0.0f, sq = 0.0f;
#pragma unroll
    for (int i = 0; i < 32; ++i) {
      float4 v = x4[i];
      buf[i] = v;
      s  += v.x + v.y + v.z + v.w;
      sq += v.x * v.x + v.y * v.y + v.z * v.z + v.w * v.w;
    }
    s  += __shfl_xor(s, 1, 32);  s  += __shfl_xor(s, 2, 32);
    sq += __shfl_xor(sq, 1, 32); sq += __shfl_xor(sq, 2, 32);
    const float mean = s * (1.0f / 512.0f);
    const float var  = sq * (1.0f / 512.0f) - mean * mean;
    const float rstd = rsqrtf(var + 1e-5f);
#pragma unroll
    for (int i = 0; i < 32; ++i) {
      const int c = q4 * 128 + i * 4;
      float4 v = buf[i];
      xnh[row * 520 + c + 0] = (_Float16)((v.x - mean) * rstd * g1[c + 0] + be1[c + 0]);
      xnh[row * 520 + c + 1] = (_Float16)((v.y - mean) * rstd * g1[c + 1] + be1[c + 1]);
      xnh[row * 520 + c + 2] = (_Float16)((v.z - mean) * rstd * g1[c + 2] + be1[c + 2]);
      xnh[row * 520 + c + 3] = (_Float16)((v.w - mean) * rstd * g1[c + 3] + be1[c + 3]);
    }
  }
  __syncthreads();

  // stage 2: scores via WMMA, transposed: S^T(16x16) = QK(16x32k) * XN^T.
  // A-frag = qkh row (head), B-frag = xnh row (reg image of A-frag of XN).
  // Waves 0..3 take one 16-row tile each; all fragment loads are b128.
  if (wv < 4) {
    const int rt = wv;
    const int m  = lane & 15;
    const int hi = lane >> 4;
    const _Float16* arow = qkh + m * 520;              // head m (zero if m>=4)
    const _Float16* brow = xnh + (rt * 16 + m) * 520;  // xn row rt*16+m
    v8f dacc = {0.f, 0.f, 0.f, 0.f, 0.f, 0.f, 0.f, 0.f};
#pragma unroll 4
    for (int kb = 0; kb < 16; ++kb) {
      v16h a = load_frag(arow + kb * 32, hi);
      v16h b = load_frag(brow + kb * 32, hi);
      dacc = __builtin_amdgcn_wmma_f32_16x16x32_f16(false, a, false, b,
                                                    (short)0, dacc, false, false);
    }
    // D: M = r + 8*hi = head, N = m = row-in-tile. Heads live in r<4, hi==0.
    if (hi == 0) {
#pragma unroll
      for (int r = 0; r < 4; ++r)
        sc[(rt * 16 + m) * 4 + r] = dacc[r] + sbl[r];
    }
  }
  __syncthreads();

  // stage 3: block-local softmax statistics per head
  if (tid < 4) {
    float m = -INFINITY;
    for (int l = 0; l < TILE; ++l) m = fmaxf(m, sc[l * 4 + tid]);
    float s = 0.0f;
    for (int l = 0; l < TILE; ++l) s += __expf(sc[l * 4 + tid] - m);
    mh[tid] = m; lh[tid] = s;
    ws[WS_MP + blk * 4 + tid] = m;
    ws[WS_LP + blk * 4 + tid] = s;
  }
  __syncthreads();
  for (int e = tid; e < 16 * TILE; e += 256) {    // pmat[16][64] (heads 0..3 live)
    const int mrow = e >> 6;
    const int l    = e & 63;
    pmat[mrow * 72 + l] =
        (mrow < 4) ? (_Float16)__expf(sc[l * 4 + mrow] - mh[mrow]) : (_Float16)0.0f;
  }
  __syncthreads();

  // stage 4: t partial = P(16x64) @ XN(64x512) via WMMA; 32 N-tiles / 8 waves.
  // A-frag = pmat row (b128 loads); B contracts over xn *rows* -> strided
  // (divergence-free scalar ds_load_u16 pairs).
  for (int nt = wv; nt < 32; nt += 8) {
    const int m  = lane & 15;
    const int hi = lane >> 4;
    v8f d = {0.f, 0.f, 0.f, 0.f, 0.f, 0.f, 0.f, 0.f};
#pragma unroll
    for (int kb = 0; kb < 2; ++kb) {
      const int kbase = kb * 32;
      v16h a = load_frag(pmat + m * 72 + kbase, hi);
      v16h b;
#pragma unroll
      for (int r = 0; r < 8; ++r) {
        const int k0 = ((r < 4) ? 0 : 16) + hi * 8 + (r & 3) * 2;
        b[2 * r]     = xnh[(kbase + k0) * 520 + nt * 16 + m];
        b[2 * r + 1] = xnh[(kbase + k0 + 1) * 520 + nt * 16 + m];
      }
      d = __builtin_amdgcn_wmma_f32_16x16x32_f16(false, a, false, b,
                                                 (short)0, d, false, false);
    }
    if (hi == 0) {                                // M=0..3 live, lanes 0..15
#pragma unroll
      for (int r = 0; r < 4; ++r)
        ws[WS_TP + (size_t)blk * 2048 + r * 512 + nt * 16 + m] = d[r];
    }
  }
}

// ---------------------------------------------------------------------------
// K2a: global softmax max/denominator merge across blocks.
// ---------------------------------------------------------------------------
__global__ void k_softmax_global(float* __restrict__ ws) {
  const int t = threadIdx.x;
  if (t < 4) {
    float M = -INFINITY;
    for (int b = 0; b < NBLK; ++b) M = fmaxf(M, ws[WS_MP + b * 4 + t]);
    float Lsum = 0.0f;
    for (int b = 0; b < NBLK; ++b)
      Lsum += ws[WS_LP + b * 4 + t] * __expf(ws[WS_MP + b * 4 + t] - M);
    ws[WS_MG + t] = M;
    ws[WS_LG + t] = Lsum;
  }
}

// ---------------------------------------------------------------------------
// K2b: merge t partials (2048 accumulators, coalesced over blocks).
// ---------------------------------------------------------------------------
__global__ __launch_bounds__(256) void k_reduce_t(float* __restrict__ ws) {
  const int e = blockIdx.x * 256 + threadIdx.x;   // 0..2047 = h*512+d
  const int h = e >> 9;
  const float M = ws[WS_MG + h];
  const float Lsum = ws[WS_LG + h];
  float acc = 0.0f;
  for (int b = 0; b < NBLK; ++b)
    acc += ws[WS_TP + (size_t)b * 2048 + e] * __expf(ws[WS_MP + b * 4 + h] - M);
  ws[WS_OPRE + e] = acc / Lsum;
}

// ---------------------------------------------------------------------------
// K3: tail on one row: V-proj of weighted-xn, out-proj, residual, LN2, MLP.
// ---------------------------------------------------------------------------
__global__ __launch_bounds__(256) void k_tail(const float* __restrict__ x,
                                              const float* __restrict__ w_in,
                                              const float* __restrict__ b_in,
                                              const float* __restrict__ w_out,
                                              const float* __restrict__ b_out,
                                              const float* __restrict__ w1,
                                              const float* __restrict__ b1,
                                              const float* __restrict__ w2,
                                              const float* __restrict__ b2,
                                              const float* __restrict__ g2,
                                              const float* __restrict__ be2,
                                              const float* __restrict__ ws,
                                              float* __restrict__ out) {
  __shared__ float sO[512], sy[512], syn[512], sh1[512];
  __shared__ float redS[256], redQ[256];
  const int t = threadIdx.x;

  // o[j] = W_v[j,:] . opre[head(j),:] + b_v[j]
  for (int j = t; j < 512; j += 256) {
    const int h = j >> 7;
    const float* wr = w_in + (size_t)(1024 + j) * 512;
    const float* op = ws + WS_OPRE + h * 512;
    float acc = b_in[1024 + j];
    for (int d = 0; d < 512; ++d) acc += wr[d] * op[d];
    sO[j] = acc;
  }
  __syncthreads();

  // y = x[0,:] + o @ w_out^T + b_out
  for (int j = t; j < 512; j += 256) {
    const float* wr = w_out + (size_t)j * 512;
    float acc = b_out[j];
    for (int i = 0; i < 512; ++i) acc += wr[i] * sO[i];
    sy[j] = x[j] + acc;
  }
  __syncthreads();

  // LN2
  {
    float a0 = sy[t], a1 = sy[t + 256];
    redS[t] = a0 + a1;
    redQ[t] = a0 * a0 + a1 * a1;
  }
  __syncthreads();
  for (int s = 128; s > 0; s >>= 1) {
    if (t < s) { redS[t] += redS[t + s]; redQ[t] += redQ[t + s]; }
    __syncthreads();
  }
  const float mean = redS[0] * (1.0f / 512.0f);
  const float var  = redQ[0] * (1.0f / 512.0f) - mean * mean;
  const float rstd = rsqrtf(var + 1e-5f);
  syn[t]       = (sy[t]       - mean) * rstd * g2[t]       + be2[t];
  syn[t + 256] = (sy[t + 256] - mean) * rstd * g2[t + 256] + be2[t + 256];
  __syncthreads();

  for (int j = t; j < 512; j += 256) {
    const float* wr = w1 + (size_t)j * 512;
    float acc = b1[j];
    for (int i = 0; i < 512; ++i) acc += wr[i] * syn[i];
    sh1[j] = fmaxf(acc, 0.0f);
  }
  __syncthreads();

  for (int j = t; j < 512; j += 256) {
    const float* wr = w2 + (size_t)j * 512;
    float acc = b2[j];
    for (int i = 0; i < 512; ++i) acc += wr[i] * sh1[i];
    out[j] = sy[j] + acc;
  }
}

// ---------------------------------------------------------------------------
extern "C" void kernel_launch(void* const* d_in, const int* in_sizes, int n_in,
                              void* d_out, int out_size, void* d_ws, size_t ws_size,
                              hipStream_t stream) {
  const float* x     = (const float*)d_in[0];
  const float* w_in  = (const float*)d_in[1];
  const float* b_in  = (const float*)d_in[2];
  const float* w_out = (const float*)d_in[3];
  const float* b_out = (const float*)d_in[4];
  const float* w1    = (const float*)d_in[5];
  const float* b1    = (const float*)d_in[6];
  const float* w2    = (const float*)d_in[7];
  const float* b2    = (const float*)d_in[8];
  const float* g1    = (const float*)d_in[9];
  const float* be1   = (const float*)d_in[10];
  const float* g2    = (const float*)d_in[11];
  const float* be2   = (const float*)d_in[12];
  float* ws  = (float*)d_ws;
  float* out = (float*)d_out;

  k_prep<<<1, 256, 0, stream>>>(x, w_in, b_in, g1, be1, ws);
  k_main<<<NBLK, 256, SMEM_BYTES, stream>>>(x, g1, be1, ws);
  k_softmax_global<<<1, 32, 0, stream>>>(ws);
  k_reduce_t<<<8, 256, 0, stream>>>(ws);
  k_tail<<<1, 256, 0, stream>>>(x, w_in, b_in, w_out, b_out, w1, b1, w2, b2,
                                g2, be2, ws, out);
}
